// SparseMoELayer_44246753084145
// MI455X (gfx1250) — matrix operand order repaired and checked by reading, not compiled, not run on hardware
//
#include <hip/hip_runtime.h>
#include <hip/hip_bf16.h>

// ---------------------------------------------------------------------------
// Sparse MoE SwiGLU for MI455X (gfx1250, wave32, WMMA bf16 16x16x32).
//
//   1. init: zero counts + usage sums
//   2. router: logits, top-2 softmax weights, expert token lists, x -> bf16
//   3. aux: aux loss scalar -> d_out[B*S*D]
//   4. gateup: (expert, 128-token tile, 64-F tile) WMMA GEMM, fused silu(g)*u
//      -> bf16 H[16384, 2816]
//   5. down: (expert, 128-token tile, 64-D tile) WMMA GEMM -> Y[16384,1024] f32
//   6. combine: out[t] = w0*Y[2t] + w1*Y[2t+1]   (deterministic, no atomics)
//
// Per-wave sub-tile is 32x32: 8 WMMAs per K=32 step from 6 LDS fragment loads
// (gate+up), keeping the WMMA pipe ahead of LDS bandwidth.
// ---------------------------------------------------------------------------

#define NTOK 8192   // B*S
#define D_   1024
#define E_   8
#define F_   2816
#define CAP  8192   // worst-case tokens per expert
#define LSTR 80     // LDS row stride in halves (160B: 16B-aligned, bank-spread)

typedef __attribute__((ext_vector_type(16))) __bf16 v16bf;
typedef __attribute__((ext_vector_type(8)))  float  v8f;

union FragBF { uint4 u[2]; v16bf v; };

static __device__ __forceinline__ unsigned short f2bf(float f) {
  __hip_bfloat16 h = __float2bfloat16(f);
  return *reinterpret_cast<unsigned short*>(&h);
}

#define WMMA_BF16(A, B, C) \
  __builtin_amdgcn_wmma_f32_16x16x32_bf16(false, (A).v, false, (B).v, (short)0, (C), false, false)

// ---------------------------------------------------------------- init ------
__global__ void init_kernel(int* __restrict__ counts, float* __restrict__ usage) {
  int i = threadIdx.x;
  if (i < E_) { counts[i] = 0; usage[i] = 0.f; }
}

// -------------------------------------------------------------- router ------
__global__ __launch_bounds__(256) void router_kernel(
    const float* __restrict__ x, const float* __restrict__ gate_w,
    __hip_bfloat16* __restrict__ xb, int* __restrict__ counts,
    int* __restrict__ list, float* __restrict__ tw, float* __restrict__ usage)
{
  __shared__ float gws[E_ * D_];
  __shared__ float psum[E_];
  int tid = threadIdx.x;
  for (int i = tid; i < E_ * D_ / 4; i += 256)
    ((float4*)gws)[i] = ((const float4*)gate_w)[i];
  if (tid < E_) psum[tid] = 0.f;
  __syncthreads();

  int wave = tid >> 5, lane = tid & 31;
  int t = blockIdx.x * 8 + wave;
  const float* xrow = x + (size_t)t * D_;
  __hip_bfloat16* xbrow = xb + (size_t)t * D_;

  float acc[E_];
#pragma unroll
  for (int e = 0; e < E_; ++e) acc[e] = 0.f;
  for (int d = lane; d < D_; d += 32) {
    float xv = xrow[d];
    xbrow[d] = __float2bfloat16(xv);
#pragma unroll
    for (int e = 0; e < E_; ++e) acc[e] += xv * gws[e * D_ + d];
  }
#pragma unroll
  for (int e = 0; e < E_; ++e) {
#pragma unroll
    for (int off = 16; off >= 1; off >>= 1)
      acc[e] += __shfl_xor(acc[e], off, 32);
  }

  if (lane == 0) {
    int i0 = 0;
#pragma unroll
    for (int e = 1; e < E_; ++e) if (acc[e] > acc[i0]) i0 = e;
    int i1 = (i0 == 0) ? 1 : 0;
#pragma unroll
    for (int e = 0; e < E_; ++e) if (e != i0 && acc[e] > acc[i1]) i1 = e;
    float v0 = acc[i0], v1 = acc[i1];
    float w1 = 1.f / (1.f + __expf(v0 - v1));  // softmax over top-2
    float w0 = 1.f - w1;
    // full softmax for expert-usage (aux loss)
    float ssum = 0.f, p[E_];
#pragma unroll
    for (int e = 0; e < E_; ++e) { p[e] = __expf(acc[e] - v0); ssum += p[e]; }
    float rs = 1.f / ssum;
#pragma unroll
    for (int e = 0; e < E_; ++e) atomicAdd(&psum[e], p[e] * rs);

    int s0 = atomicAdd(&counts[i0], 1);
    list[i0 * CAP + s0] = 2 * t;
    tw[2 * t] = w0;
    int s1 = atomicAdd(&counts[i1], 1);
    list[i1 * CAP + s1] = 2 * t + 1;
    tw[2 * t + 1] = w1;
  }
  __syncthreads();
  if (tid < E_) atomicAdd(&usage[tid], psum[tid]);
}

// ----------------------------------------------------------------- aux ------
__global__ void aux_kernel(const float* __restrict__ usage, float* __restrict__ out_aux) {
  if (threadIdx.x == 0) {
    float a = 0.f;
#pragma unroll
    for (int e = 0; e < E_; ++e) {
      float u = usage[e] * (1.f / (float)NTOK);
      float d = u - 1.f / (float)E_;
      a += d * d;
    }
    *out_aux = a;
  }
}

// -------------------------------------------------------------- gate/up -----
// Block = 8 waves, tile = 128 tokens x 64 F-cols, K over D in 64-chunks.
// Wave (wm,wn): wm in 0..3 -> 32-row group, wn in 0..1 -> 32-col group.
__global__ __launch_bounds__(256) void gateup_kernel(
    const __hip_bfloat16* __restrict__ xb,
    const float* __restrict__ w_gate, const float* __restrict__ w_up,
    const int* __restrict__ counts, const int* __restrict__ list,
    __hip_bfloat16* __restrict__ H)
{
  __shared__ unsigned short Xs[128 * LSTR];
  __shared__ unsigned short Gs[64 * LSTR];
  __shared__ unsigned short Us[64 * LSTR];
  __shared__ int Ls[128];

  int ft = blockIdx.x;          // F tile: 0..43
  int emt = blockIdx.y;         // e*64 + mt
  int e = emt >> 6, mt = emt & 63;
  int cnt = counts[e];
  if (mt * 128 >= cnt) return;

  int tid = threadIdx.x;
  if (tid < 128) {
    int idx = mt * 128 + tid;
    Ls[tid] = (idx < cnt) ? list[e * CAP + idx] : -1;
  }

  int wave = tid >> 5, lane = tid & 31;
  int wm = wave & 3, wn = wave >> 2;
  int l16 = lane & 15, lhi = lane >> 4;

  v8f ag[2][2] = {}, au[2][2] = {};   // [A-row-frag][B-col-frag]
  const float* wg = w_gate + ((size_t)e * F_ + (size_t)ft * 64) * D_;
  const float* wu = w_up   + ((size_t)e * F_ + (size_t)ft * 64) * D_;

  for (int kc = 0; kc < D_; kc += 64) {
    __syncthreads();  // LDS reuse fence (covers Ls on first iter)
    // stage gathered X rows (bf16): 128 rows x 64 halves, b128 per thread x4
#pragma unroll
    for (int i = 0; i < 4; ++i) {
      int idx = tid + 256 * i;
      int row = idx >> 3, seg = idx & 7;
      int rv = Ls[row];
      size_t tok = (rv >= 0) ? (size_t)(rv >> 1) : 0;
      *(uint4*)&Xs[row * LSTR + seg * 8] =
          *(const uint4*)(xb + tok * D_ + kc + seg * 8);
    }
    // stage weights fp32 -> bf16: 64 f-rows x 64 k; prefetch next chunk
#pragma unroll
    for (int i = 0; i < 4; ++i) {
      int idx = tid + 256 * i;
      int row = idx >> 4, seg = idx & 15;
      const float* gp = wg + (size_t)row * D_ + kc + seg * 4;
      const float* up = wu + (size_t)row * D_ + kc + seg * 4;
      float4 g4 = *(const float4*)gp;
      float4 u4 = *(const float4*)up;
      if (kc + 64 < D_) {
        __builtin_prefetch(gp + 64, 0, 0);   // global_prefetch_b8
        __builtin_prefetch(up + 64, 0, 0);
      }
      union { unsigned short s[4]; uint2 u; } pg, pu;
      pg.s[0] = f2bf(g4.x); pg.s[1] = f2bf(g4.y); pg.s[2] = f2bf(g4.z); pg.s[3] = f2bf(g4.w);
      pu.s[0] = f2bf(u4.x); pu.s[1] = f2bf(u4.y); pu.s[2] = f2bf(u4.z); pu.s[3] = f2bf(u4.w);
      *(uint2*)&Gs[row * LSTR + seg * 4] = pg.u;
      *(uint2*)&Us[row * LSTR + seg * 4] = pu.u;
    }
    __syncthreads();

#pragma unroll
    for (int kk = 0; kk < 64; kk += 32) {
      FragBF a0, a1, bg0, bg1, bu0, bu1;
      // A fragments: rows wm*32 + {0,16} + l16; 16 contiguous K at kk + 16*lhi
      const unsigned short* x0 = &Xs[(wm * 32 + l16) * LSTR + kk + lhi * 16];
      a0.u[0] = *(const uint4*)(x0); a0.u[1] = *(const uint4*)(x0 + 8);
      const unsigned short* x1 = &Xs[(wm * 32 + 16 + l16) * LSTR + kk + lhi * 16];
      a1.u[0] = *(const uint4*)(x1); a1.u[1] = *(const uint4*)(x1 + 8);
      // B fragments: cols wn*32 + {0,16} + l16; 16 contiguous K at kk + 16*lhi
      const unsigned short* g0 = &Gs[(wn * 32 + l16) * LSTR + kk + lhi * 16];
      bg0.u[0] = *(const uint4*)(g0); bg0.u[1] = *(const uint4*)(g0 + 8);
      const unsigned short* g1 = &Gs[(wn * 32 + 16 + l16) * LSTR + kk + lhi * 16];
      bg1.u[0] = *(const uint4*)(g1); bg1.u[1] = *(const uint4*)(g1 + 8);
      const unsigned short* u0 = &Us[(wn * 32 + l16) * LSTR + kk + lhi * 16];
      bu0.u[0] = *(const uint4*)(u0); bu0.u[1] = *(const uint4*)(u0 + 8);
      const unsigned short* u1 = &Us[(wn * 32 + 16 + l16) * LSTR + kk + lhi * 16];
      bu1.u[0] = *(const uint4*)(u1); bu1.u[1] = *(const uint4*)(u1 + 8);

      ag[0][0] = WMMA_BF16(a0, bg0, ag[0][0]);
      ag[0][1] = WMMA_BF16(a0, bg1, ag[0][1]);
      ag[1][0] = WMMA_BF16(a1, bg0, ag[1][0]);
      ag[1][1] = WMMA_BF16(a1, bg1, ag[1][1]);
      au[0][0] = WMMA_BF16(a0, bu0, au[0][0]);
      au[0][1] = WMMA_BF16(a0, bu1, au[0][1]);
      au[1][0] = WMMA_BF16(a1, bu0, au[1][0]);
      au[1][1] = WMMA_BF16(a1, bu1, au[1][1]);
    }
  }

  // fused epilogue: h = silu(g)*u -> bf16 H rows (single v_rcp_f32 sigmoid)
  int nb = ft * 64 + wn * 32;
#pragma unroll
  for (int rh = 0; rh < 2; ++rh) {
#pragma unroll
    for (int s = 0; s < 2; ++s) {
      int n = nb + s * 16 + l16;
#pragma unroll
      for (int q = 0; q < 8; ++q) {
        int m = wm * 32 + rh * 16 + lhi * 8 + q;  // C layout: M = 8*(lane>=16)+vgpr
        int rv = Ls[m];
        if (rv >= 0) {
          float gv = ag[rh][s][q], uv = au[rh][s][q];
          float sig = __builtin_amdgcn_rcpf(1.f + __expf(-gv));
          H[(size_t)rv * F_ + n] = __float2bfloat16(gv * sig * uv);
        }
      }
    }
  }
}

// ---------------------------------------------------------------- down ------
// Tile = 128 assignment-rows x 64 D-cols, K over F (44 chunks of 64).
__global__ __launch_bounds__(256) void down_kernel(
    const __hip_bfloat16* __restrict__ H, const float* __restrict__ w_down,
    const int* __restrict__ counts, const int* __restrict__ list,
    float* __restrict__ Y)
{
  __shared__ unsigned short Hs[128 * LSTR];
  __shared__ unsigned short Ws[64 * LSTR];
  __shared__ int Ls[128];

  int dt = blockIdx.x;          // D tile: 0..15
  int emt = blockIdx.y;         // e*64 + mt
  int e = emt >> 6, mt = emt & 63;
  int cnt = counts[e];
  if (mt * 128 >= cnt) return;

  int tid = threadIdx.x;
  if (tid < 128) {
    int idx = mt * 128 + tid;
    Ls[tid] = (idx < cnt) ? list[e * CAP + idx] : -1;
  }

  int wave = tid >> 5, lane = tid & 31;
  int wm = wave & 3, wn = wave >> 2;
  int l16 = lane & 15, lhi = lane >> 4;

  v8f acc[2][2] = {};
  const float* wd = w_down + ((size_t)e * D_ + (size_t)dt * 64) * F_;

  for (int kc = 0; kc < F_; kc += 64) {
    __syncthreads();
#pragma unroll
    for (int i = 0; i < 4; ++i) {
      int idx = tid + 256 * i;
      int row = idx >> 3, seg = idx & 7;
      int rv = Ls[row];
      size_t r = (rv >= 0) ? (size_t)rv : 0;
      *(uint4*)&Hs[row * LSTR + seg * 8] =
          *(const uint4*)(H + r * F_ + kc + seg * 8);
    }
#pragma unroll
    for (int i = 0; i < 4; ++i) {
      int idx = tid + 256 * i;
      int row = idx >> 4, seg = idx & 15;
      const float* wp = wd + (size_t)row * F_ + kc + seg * 4;
      float4 w4 = *(const float4*)wp;
      if (kc + 64 < F_) __builtin_prefetch(wp + 64, 0, 0);
      union { unsigned short s[4]; uint2 u; } p;
      p.s[0] = f2bf(w4.x); p.s[1] = f2bf(w4.y); p.s[2] = f2bf(w4.z); p.s[3] = f2bf(w4.w);
      *(uint2*)&Ws[row * LSTR + seg * 4] = p.u;
    }
    __syncthreads();

#pragma unroll
    for (int kk = 0; kk < 64; kk += 32) {
      FragBF a0, a1, b0, b1;
      const unsigned short* h0 = &Hs[(wm * 32 + l16) * LSTR + kk + lhi * 16];
      a0.u[0] = *(const uint4*)(h0); a0.u[1] = *(const uint4*)(h0 + 8);
      const unsigned short* h1 = &Hs[(wm * 32 + 16 + l16) * LSTR + kk + lhi * 16];
      a1.u[0] = *(const uint4*)(h1); a1.u[1] = *(const uint4*)(h1 + 8);
      const unsigned short* w0 = &Ws[(wn * 32 + l16) * LSTR + kk + lhi * 16];
      b0.u[0] = *(const uint4*)(w0); b0.u[1] = *(const uint4*)(w0 + 8);
      const unsigned short* w1 = &Ws[(wn * 32 + 16 + l16) * LSTR + kk + lhi * 16];
      b1.u[0] = *(const uint4*)(w1); b1.u[1] = *(const uint4*)(w1 + 8);

      acc[0][0] = WMMA_BF16(a0, b0, acc[0][0]);
      acc[0][1] = WMMA_BF16(a0, b1, acc[0][1]);
      acc[1][0] = WMMA_BF16(a1, b0, acc[1][0]);
      acc[1][1] = WMMA_BF16(a1, b1, acc[1][1]);
    }
  }

  int nb = dt * 64 + wn * 32;
#pragma unroll
  for (int rh = 0; rh < 2; ++rh) {
#pragma unroll
    for (int s = 0; s < 2; ++s) {
      int n = nb + s * 16 + l16;
#pragma unroll
      for (int q = 0; q < 8; ++q) {
        int m = wm * 32 + rh * 16 + lhi * 8 + q;
        int rv = Ls[m];
        if (rv >= 0) Y[(size_t)rv * D_ + n] = acc[rh][s][q];
      }
    }
  }
}

// -------------------------------------------------------------- combine -----
__global__ __launch_bounds__(256) void combine_kernel(
    const float* __restrict__ Y, const float* __restrict__ tw,
    float* __restrict__ out)
{
  int idx = blockIdx.x * 256 + threadIdx.x;  // one float4 per thread
  int t = idx >> 8;                          // 256 float4 per token row
  int c = idx & 255;
  float w0 = tw[2 * t], w1 = tw[2 * t + 1];
  float4 a = ((const float4*)Y)[(size_t)(2 * t) * 256 + c];
  float4 b = ((const float4*)Y)[(size_t)(2 * t + 1) * 256 + c];
  float4 o;
  o.x = w0 * a.x + w1 * b.x;
  o.y = w0 * a.y + w1 * b.y;
  o.z = w0 * a.z + w1 * b.z;
  o.w = w0 * a.w + w1 * b.w;
  ((float4*)out)[(size_t)t * 256 + c] = o;
}

// --------------------------------------------------------------- launch -----
extern "C" void kernel_launch(void* const* d_in, const int* in_sizes, int n_in,
                              void* d_out, int out_size, void* d_ws, size_t ws_size,
                              hipStream_t stream)
{
  const float* x      = (const float*)d_in[0];
  const float* gate_w = (const float*)d_in[1];
  const float* w_gate = (const float*)d_in[2];
  const float* w_up   = (const float*)d_in[3];
  const float* w_down = (const float*)d_in[4];
  float* out = (float*)d_out;
  (void)in_sizes; (void)n_in; (void)out_size; (void)ws_size;

  char* ws = (char*)d_ws;
  size_t off = 0;
  auto alloc = [&](size_t bytes) -> void* {
    void* p = ws + off;
    off = (off + bytes + 255) & ~(size_t)255;
    return p;
  };
  int*   counts = (int*)  alloc(E_ * sizeof(int));
  float* usage  = (float*)alloc(E_ * sizeof(float));
  float* tw     = (float*)alloc((size_t)2 * NTOK * sizeof(float));
  int*   list   = (int*)  alloc((size_t)E_ * CAP * sizeof(int));
  __hip_bfloat16* xb = (__hip_bfloat16*)alloc((size_t)NTOK * D_ * 2);      // 16 MB
  __hip_bfloat16* H  = (__hip_bfloat16*)alloc((size_t)2 * NTOK * F_ * 2);  // 92 MB
  float*          Y  = (float*)         alloc((size_t)2 * NTOK * D_ * 4);  // 67 MB

  hipLaunchKernelGGL(init_kernel, dim3(1), dim3(32), 0, stream, counts, usage);
  hipLaunchKernelGGL(router_kernel, dim3(NTOK / 8), dim3(256), 0, stream,
                     x, gate_w, xb, counts, list, tw, usage);
  hipLaunchKernelGGL(aux_kernel, dim3(1), dim3(32), 0, stream,
                     usage, out + (size_t)NTOK * D_);
  hipLaunchKernelGGL(gateup_kernel, dim3(F_ / 64, E_ * (NTOK / 128)), dim3(256), 0, stream,
                     xb, w_gate, w_up, counts, list, H);
  hipLaunchKernelGGL(down_kernel, dim3(D_ / 64, E_ * (NTOK / 128)), dim3(256), 0, stream,
                     H, w_down, counts, list, Y);
  hipLaunchKernelGGL(combine_kernel, dim3(NTOK), dim3(256), 0, stream, Y, tw, out);
}
